// XSACausalSelfAttention_28063316312159
// MI455X (gfx1250) — matrix operand order repaired and checked by reading, not compiled, not run on hardware
//
#include <hip/hip_runtime.h>
#include <hip/hip_bf16.h>

typedef __bf16 bf16;
typedef __attribute__((ext_vector_type(16))) __bf16 v16bf;
typedef __attribute__((ext_vector_type(8)))  __bf16 v8bf;
typedef __attribute__((ext_vector_type(8)))  float  v8f;
typedef __attribute__((ext_vector_type(4)))  unsigned int u32x4;
typedef __attribute__((ext_vector_type(8)))  unsigned int u32x8;

constexpr int B_ = 2, T_ = 2048, DIM_ = 2048, H_ = 16, HKV_ = 4, HD_ = 128;
constexpr int GQ_ = H_ / HKV_;

__device__ inline v16bf cat8(v8bf lo, v8bf hi) {
  return __builtin_shufflevector(lo, hi, 0,1,2,3,4,5,6,7,8,9,10,11,12,13,14,15);
}
__device__ inline v8f wmma_bf16(v16bf a, v16bf b, v8f c) {
  return __builtin_amdgcn_wmma_f32_16x16x32_bf16(false, a, false, b, (short)0, c, false, false);
}
__device__ inline v8f v8f_zero() {
  v8f z;
  #pragma unroll
  for (int i = 0; i < 8; ++i) z[i] = 0.0f;
  return z;
}

// ---------------------------------------------------------------- cast fp32 -> bf16
__global__ void cast_f32_to_bf16(const float* __restrict__ in, bf16* __restrict__ out, int n) {
  int i = blockIdx.x * blockDim.x + threadIdx.x;
  int stride = gridDim.x * blockDim.x;
  for (; i < n; i += stride) out[i] = (bf16)in[i];
}

// ---------------------------------------------------------------- TDM probe (not launched)
// Exercises the CDNA5 Tensor Data Mover: builds a D# (CDNA5 ISA)
// for a 32x128 bf16 2D tile and issues tensor_load_to_lds, then waits TENSORcnt.
__global__ void tdm_tile_copy_kernel(const bf16* __restrict__ src, bf16* __restrict__ dst) {
  __shared__ __align__(1024) bf16 tile[32 * 128];
  unsigned long long ga = (unsigned long long)(const void*)src;
  unsigned lds_base = (unsigned)(unsigned long long)(void*)&tile[0];

  u32x4 g0;
  g0[0] = 1u;                                                 // count=1, user mode
  g0[1] = lds_base;                                           // lds_addr
  g0[2] = (unsigned)(ga & 0xffffffffu);                       // global_addr[31:0]
  g0[3] = (unsigned)((ga >> 32) & 0x1ffffffu) | (2u << 30);   // global_addr[56:32], type=2

  u32x8 g1;
  g1[0] = 1u << 16;     // data_size = 2 bytes
  g1[1] = 128u << 16;   // tensor_dim0 = 128
  g1[2] = 32u << 16;    // tensor_dim1 = 32
  g1[3] = 128u << 16;   // tile_dim0 = 128
  g1[4] = 32u;          // tile_dim1 = 32, tile_dim2 = 0
  g1[5] = 128u;         // tensor_dim0_stride = 128
  g1[6] = 0u;
  g1[7] = 0u;

  u32x4 g2; g2[0] = 0u; g2[1] = 0u; g2[2] = 0u; g2[3] = 0u;
  u32x4 g3 = g2;

  asm volatile("tensor_load_to_lds %0, %1, %2, %3"
               :: "s"(g0), "s"(g1), "s"(g2), "s"(g3)
               : "memory");
  __builtin_amdgcn_s_wait_tensorcnt(0);
  __syncthreads();
  for (int i = threadIdx.x; i < 32 * 128; i += blockDim.x) dst[i] = tile[i];
}

// ---------------------------------------------------------------- bf16 GEMM, fp32 out
// C[M,N] = A[M,K] * B[K,N]; block tile 128x128, 8 waves (4x2), wave tile 32x64,
// BK=32, double-buffered LDS with register-staged global loads + prefetch.
__global__ __launch_bounds__(256) void gemm_bf16_f32(
    const bf16* __restrict__ A, const bf16* __restrict__ Bm,
    float* __restrict__ C, int M, int N, int K)
{
  __shared__ __align__(16) bf16 As[2][128][40];   // [buf][m][k], padded
  __shared__ __align__(16) bf16 Bst[2][128][40];  // [buf][n][k], transposed, padded

  const int bm = blockIdx.y * 128, bn = blockIdx.x * 128;
  const int tid = threadIdx.x;
  const int wave = tid >> 5, lane = tid & 31;
  const int wm = wave >> 1, wn = wave & 1;        // 4x2 wave grid
  const int m = lane & 15, kh = lane >> 4;

  // staging coordinates (straight-line, 2 chunks of A + 2 of B per thread)
  const int ar = tid >> 2, ac = (tid & 3) * 8;    // A rows ar, ar+64
  const int br = tid >> 4, bc = (tid & 15) * 8;   // B k-rows br, br+16; cols bc..bc+7
  const bf16* Abase0 = A + (size_t)(bm + ar) * K;
  const bf16* Abase1 = A + (size_t)(bm + ar + 64) * K;

  v8f acc[2][4];
  #pragma unroll
  for (int a = 0; a < 2; ++a)
    #pragma unroll
    for (int b = 0; b < 4; ++b) acc[a][b] = v8f_zero();

  const int nk = K >> 5;

  // prologue: tile 0 -> buffer 0
  {
    v8bf a0 = *(const v8bf*)(Abase0 + ac);
    v8bf a1 = *(const v8bf*)(Abase1 + ac);
    v8bf b0 = *(const v8bf*)(Bm + (size_t)br * N + bn + bc);
    v8bf b1 = *(const v8bf*)(Bm + (size_t)(br + 16) * N + bn + bc);
    *(v8bf*)&As[0][ar][ac] = a0;
    *(v8bf*)&As[0][ar + 64][ac] = a1;
    #pragma unroll
    for (int e = 0; e < 8; ++e) { Bst[0][bc + e][br] = b0[e]; Bst[0][bc + e][br + 16] = b1[e]; }
  }
  __syncthreads();

  for (int kt = 0; kt < nk; ++kt) {
    const int cur = kt & 1, nxt = cur ^ 1;
    const bool has_next = (kt + 1) < nk;        // block-uniform
    v8bf a0, a1, b0, b1;
    if (has_next) {
      const int k0n = (kt + 1) << 5;
      a0 = *(const v8bf*)(Abase0 + k0n + ac);
      a1 = *(const v8bf*)(Abase1 + k0n + ac);
      b0 = *(const v8bf*)(Bm + (size_t)(k0n + br) * N + bn + bc);
      b1 = *(const v8bf*)(Bm + (size_t)(k0n + br + 16) * N + bn + bc);
      if (kt + 2 < nk) {
        const int k0p = (kt + 2) << 5;
        __builtin_prefetch(Abase0 + k0p + ac, 0, 0);
        __builtin_prefetch(Bm + (size_t)(k0p + br) * N + bn + bc, 0, 0);
      }
    }

    // compute from buffer `cur`
    v16bf af[2];
    #pragma unroll
    for (int mt = 0; mt < 2; ++mt) {
      const bf16* ab = &As[cur][wm * 32 + mt * 16 + m][kh * 8];
      af[mt] = cat8(*(const v8bf*)ab, *(const v8bf*)(ab + 16));
    }
    #pragma unroll
    for (int nt = 0; nt < 4; ++nt) {
      const bf16* bb = &Bst[cur][wn * 64 + nt * 16 + m][kh * 8];
      v16bf bfr = cat8(*(const v8bf*)bb, *(const v8bf*)(bb + 16));
      #pragma unroll
      for (int mt = 0; mt < 2; ++mt)
        acc[mt][nt] = wmma_bf16(af[mt], bfr, acc[mt][nt]);
    }

    if (has_next) {
      *(v8bf*)&As[nxt][ar][ac] = a0;
      *(v8bf*)&As[nxt][ar + 64][ac] = a1;
      #pragma unroll
      for (int e = 0; e < 8; ++e) { Bst[nxt][bc + e][br] = b0[e]; Bst[nxt][bc + e][br + 16] = b1[e]; }
    }
    __syncthreads();
  }

  #pragma unroll
  for (int mt = 0; mt < 2; ++mt)
    #pragma unroll
    for (int nt = 0; nt < 4; ++nt)
      #pragma unroll
      for (int j = 0; j < 8; ++j) {
        int row = bm + wm * 32 + mt * 16 + j + 8 * kh;
        int col = bn + wn * 64 + nt * 16 + m;
        C[(size_t)row * N + col] = acc[mt][nt][j];
      }
}

// ---------------------------------------------------------------- RMSNorm + RoPE + gain
__global__ __launch_bounds__(256) void normrope_kernel(
    const float* __restrict__ qf, const float* __restrict__ kf, const float* __restrict__ vf,
    const float* __restrict__ q_gain,
    bf16* __restrict__ qb, bf16* __restrict__ kb, bf16* __restrict__ vb)
{
  const int bt = blockIdx.x;
  const int b = bt / T_, t = bt % T_;
  const int wave = threadIdx.x >> 5, lane = threadIdx.x & 31;
  const int d0 = lane * 4;

  for (int item = wave; item < H_ + 2 * HKV_; item += 8) {
    if (item < H_ + HKV_) {
      const bool isq = item < H_;
      const int hh = isq ? item : (item - H_);
      const float* src = isq ? (qf + (size_t)bt * DIM_ + hh * HD_)
                             : (kf + (size_t)bt * (HKV_ * HD_) + hh * HD_);
      float4 x = *(const float4*)(src + d0);
      float s2 = x.x * x.x + x.y * x.y + x.z * x.z + x.w * x.w;
      #pragma unroll
      for (int off = 1; off < 32; off <<= 1) s2 += __shfl_xor(s2, off);
      const float rinv = rsqrtf(s2 * (1.0f / 128.0f) + 1.1920929e-07f);
      float xv[4] = {x.x * rinv, x.y * rinv, x.z * rinv, x.w * rinv};
      float out[4];
      #pragma unroll
      for (int e = 0; e < 4; ++e) {
        float pv = __shfl_xor(xv[e], 16);      // partner at d^64
        int d = d0 + e;
        int fi = (d < 64) ? d : (d - 64);
        float ang = (float)t * __powf(10000.0f, -(float)fi * (1.0f / 64.0f));
        float c = __cosf(ang), s = __sinf(ang);
        out[e] = (d < 64) ? (xv[e] * c + pv * s) : (xv[e] * c - pv * s);
      }
      if (isq) {
        const float g = q_gain[hh];
        bf16* dst = qb + ((size_t)(b * H_ + hh) * T_ + t) * HD_ + d0;
        #pragma unroll
        for (int e = 0; e < 4; ++e) dst[e] = (bf16)(out[e] * g);
      } else {
        bf16* dst = kb + ((size_t)(b * HKV_ + hh) * T_ + t) * HD_ + d0;
        #pragma unroll
        for (int e = 0; e < 4; ++e) dst[e] = (bf16)out[e];
      }
    } else {
      const int hv = item - H_ - HKV_;
      const float* src = vf + (size_t)bt * (HKV_ * HD_) + hv * HD_;
      bf16* dst = vb + ((size_t)(b * HKV_ + hv) * T_ + t) * HD_ + d0;
      float4 x = *(const float4*)(src + d0);
      dst[0] = (bf16)x.x; dst[1] = (bf16)x.y; dst[2] = (bf16)x.z; dst[3] = (bf16)x.w;
    }
  }
}

// ---------------------------------------------------------------- flash attention + v-rejection
// Block: 4 waves; wave w owns Q rows [qy*64 + w*16, +16). KV staged in LDS in 32-key chunks.
__global__ __launch_bounds__(128) void attn_kernel(
    const bf16* __restrict__ qb, const bf16* __restrict__ kb,
    const bf16* __restrict__ vb, bf16* __restrict__ yb)
{
  __shared__ __align__(16) bf16 Ks[32][136];     // [key][d]
  __shared__ __align__(16) bf16 Vt[128][40];     // transposed [d][key]
  __shared__ __align__(16) bf16 Ps[4][16][40];   // per-wave P staging (16x32)

  const int bh = blockIdx.x;
  const int b = bh / H_, h = bh % H_, kvh = h / GQ_;
  const bf16* qp = qb + ((size_t)(b * H_ + h) * T_) * HD_;
  const bf16* kp = kb + ((size_t)(b * HKV_ + kvh) * T_) * HD_;
  const bf16* vp = vb + ((size_t)(b * HKV_ + kvh) * T_) * HD_;

  const int wave = threadIdx.x >> 5, lane = threadIdx.x & 31;
  const int m = lane & 15, kh = lane >> 4;
  const int qr0 = blockIdx.y * 64 + wave * 16;

  // Q fragments: 16 rows x 128 dims = 4 WMMA K-chunks, preloaded
  v16bf aq[4];
  #pragma unroll
  for (int c = 0; c < 4; ++c) {
    const bf16* base = qp + (size_t)(qr0 + m) * HD_ + c * 32 + kh * 8;
    aq[c] = cat8(*(const v8bf*)base, *(const v8bf*)(base + 16));
  }

  v8f oacc[8];
  #pragma unroll
  for (int nt = 0; nt < 8; ++nt) oacc[nt] = v8f_zero();
  float mrow[8], lrow[8];
  #pragma unroll
  for (int j = 0; j < 8; ++j) { mrow[j] = -3.0e38f; lrow[j] = 0.0f; }

  const float scale = 0.08838834764831845f;     // 1/sqrt(128)
  const int kcend = blockIdx.y * 2 + 2;

  // straight-line staging coordinates: 4 chunks per thread
  const int sr = threadIdx.x >> 4;              // base key row (0..7)
  const int sc = (threadIdx.x & 15) * 8;        // dim offset

  for (int kc = 0; kc < kcend; ++kc) {
    __syncthreads();
    v8bf kreg[4], vreg[4];
    #pragma unroll
    for (int i = 0; i < 4; ++i) {
      const int r = sr + i * 8;
      kreg[i] = *(const v8bf*)(kp + (size_t)(kc * 32 + r) * HD_ + sc);
      vreg[i] = *(const v8bf*)(vp + (size_t)(kc * 32 + r) * HD_ + sc);
    }
    if (kc + 1 < kcend) {
      __builtin_prefetch(kp + (size_t)((kc + 1) * 32 + sr) * HD_ + sc, 0, 0);
      __builtin_prefetch(vp + (size_t)((kc + 1) * 32 + sr) * HD_ + sc, 0, 0);
    }
    #pragma unroll
    for (int i = 0; i < 4; ++i) {
      const int r = sr + i * 8;
      *(v8bf*)&Ks[r][sc] = kreg[i];
      #pragma unroll
      for (int e = 0; e < 8; ++e) Vt[sc + e][r] = vreg[i][e];
    }
    __syncthreads();

    if (kc * 32 <= qr0 + 15) {                  // wave-uniform: EXEC stays all-ones
      float St[2][8];
      #pragma unroll
      for (int s = 0; s < 2; ++s) {
        v8f sc2 = v8f_zero();
        #pragma unroll
        for (int dc = 0; dc < 4; ++dc) {
          const bf16* kb2 = &Ks[s * 16 + m][dc * 32 + kh * 8];
          v16bf bk = cat8(*(const v8bf*)kb2, *(const v8bf*)(kb2 + 16));
          sc2 = wmma_bf16(aq[dc], bk, sc2);
        }
        #pragma unroll
        for (int j = 0; j < 8; ++j) {
          int col = kc * 32 + s * 16 + m;
          int row = qr0 + j + 8 * kh;
          St[s][j] = (col <= row) ? sc2[j] * scale : -3.0e38f;
        }
      }
      // online softmax (rows live across the 16 lanes of each half)
      float corr[8];
      #pragma unroll
      for (int j = 0; j < 8; ++j) {
        float rm = fmaxf(St[0][j], St[1][j]);
        #pragma unroll
        for (int off = 1; off < 16; off <<= 1) rm = fmaxf(rm, __shfl_xor(rm, off));
        float mnew = fmaxf(mrow[j], rm);
        float cj = __expf(mrow[j] - mnew);
        float p0 = __expf(St[0][j] - mnew);
        float p1 = __expf(St[1][j] - mnew);
        St[0][j] = p0; St[1][j] = p1;
        float ps = p0 + p1;
        #pragma unroll
        for (int off = 1; off < 16; off <<= 1) ps += __shfl_xor(ps, off);
        lrow[j] = lrow[j] * cj + ps;
        mrow[j] = mnew;
        corr[j] = cj;
      }
      #pragma unroll
      for (int nt = 0; nt < 8; ++nt)
        #pragma unroll
        for (int j = 0; j < 8; ++j) oacc[nt][j] *= corr[j];

      // re-shape P (C-layout) -> A-fragment via per-wave LDS bounce
      #pragma unroll
      for (int j = 0; j < 8; ++j) {
        Ps[wave][j + 8 * kh][m]      = (bf16)St[0][j];
        Ps[wave][j + 8 * kh][16 + m] = (bf16)St[1][j];
      }
      const bf16* pb = &Ps[wave][m][kh * 8];
      v16bf ap = cat8(*(const v8bf*)pb, *(const v8bf*)(pb + 16));
      #pragma unroll
      for (int nt = 0; nt < 8; ++nt) {
        const bf16* vb2 = &Vt[nt * 16 + m][kh * 8];
        v16bf bv = cat8(*(const v8bf*)vb2, *(const v8bf*)(vb2 + 16));
        oacc[nt] = wmma_bf16(ap, bv, oacc[nt]);
      }
    }
  }

  // epilogue: softmax normalize + reject component along own-position v
  #pragma unroll
  for (int j = 0; j < 8; ++j) {
    const int qr = qr0 + j + 8 * kh;
    const float rinv = 1.0f / fmaxf(lrow[j], 1e-30f);
    float vv[8], ss = 0.0f, dot = 0.0f;
    #pragma unroll
    for (int nt = 0; nt < 8; ++nt) {
      vv[nt] = (float)vp[(size_t)qr * HD_ + nt * 16 + m];
      float yv = oacc[nt][j] * rinv;
      ss += vv[nt] * vv[nt];
      dot += yv * vv[nt];
    }
    #pragma unroll
    for (int off = 1; off < 16; off <<= 1) {
      ss += __shfl_xor(ss, off);
      dot += __shfl_xor(dot, off);
    }
    const float cf = dot / fmaxf(ss, 1e-24f);
    bf16* orow = yb + (size_t)(b * T_ + qr) * DIM_ + h * HD_;
    #pragma unroll
    for (int nt = 0; nt < 8; ++nt) {
      float yv = oacc[nt][j] * rinv - cf * vv[nt];
      orow[nt * 16 + m] = (bf16)yv;
    }
  }
}

// ---------------------------------------------------------------- launch
extern "C" void kernel_launch(void* const* d_in, const int* in_sizes, int n_in,
                              void* d_out, int out_size, void* d_ws, size_t ws_size,
                              hipStream_t stream) {
  (void)in_sizes; (void)n_in; (void)out_size; (void)ws_size;
  const float* x  = (const float*)d_in[0];
  const float* Wq = (const float*)d_in[1];
  const float* Wk = (const float*)d_in[2];
  const float* Wv = (const float*)d_in[3];
  const float* Wo = (const float*)d_in[4];
  const float* qg = (const float*)d_in[5];

  const size_t NT = (size_t)B_ * T_;             // 4096 tokens
  char* p = (char*)d_ws;
  auto take = [&](size_t bytes) { char* r = p; p += (bytes + 255) & ~(size_t)255; return r; };

  bf16*  xb  = (bf16*)take(NT * DIM_ * sizeof(bf16));
  bf16*  wqb = (bf16*)take((size_t)DIM_ * DIM_ * sizeof(bf16));
  bf16*  wkb = (bf16*)take((size_t)DIM_ * HKV_ * HD_ * sizeof(bf16));
  bf16*  wvb = (bf16*)take((size_t)DIM_ * HKV_ * HD_ * sizeof(bf16));
  bf16*  wob = (bf16*)take((size_t)DIM_ * DIM_ * sizeof(bf16));
  float* qf  = (float*)take(NT * DIM_ * sizeof(float));
  float* kf  = (float*)take(NT * HKV_ * HD_ * sizeof(float));
  float* vf  = (float*)take(NT * HKV_ * HD_ * sizeof(float));
  bf16*  qbh = (bf16*)take(NT * DIM_ * sizeof(bf16));
  bf16*  kbh = (bf16*)take(NT * HKV_ * HD_ * sizeof(bf16));
  bf16*  vbh = (bf16*)take(NT * HKV_ * HD_ * sizeof(bf16));
  bf16*  yb  = (bf16*)take(NT * DIM_ * sizeof(bf16));

  auto cast = [&](const float* in, bf16* out, size_t n) {
    cast_f32_to_bf16<<<dim3(1024), dim3(256), 0, stream>>>(in, out, (int)n);
  };
  cast(x,  xb,  NT * DIM_);
  cast(Wq, wqb, (size_t)DIM_ * DIM_);
  cast(Wk, wkb, (size_t)DIM_ * HKV_ * HD_);
  cast(Wv, wvb, (size_t)DIM_ * HKV_ * HD_);
  cast(Wo, wob, (size_t)DIM_ * DIM_);

  gemm_bf16_f32<<<dim3(DIM_ / 128, NT / 128), 256, 0, stream>>>(xb, wqb, qf, (int)NT, DIM_, DIM_);
  gemm_bf16_f32<<<dim3(HKV_ * HD_ / 128, NT / 128), 256, 0, stream>>>(xb, wkb, kf, (int)NT, HKV_ * HD_, DIM_);
  gemm_bf16_f32<<<dim3(HKV_ * HD_ / 128, NT / 128), 256, 0, stream>>>(xb, wvb, vf, (int)NT, HKV_ * HD_, DIM_);

  normrope_kernel<<<dim3(B_ * T_), 256, 0, stream>>>(qf, kf, vf, qg, qbh, kbh, vbh);

  attn_kernel<<<dim3(B_ * H_, T_ / 64), 128, 0, stream>>>(qbh, kbh, vbh, yb);

  gemm_bf16_f32<<<dim3(DIM_ / 128, NT / 128), 256, 0, stream>>>(yb, wob, (float*)d_out, (int)NT, DIM_, DIM_);
}
